// ConvLTVFilterGenerator_80324478370382
// MI455X (gfx1250) — matrix operand order repaired and checked by reading, not compiled, not run on hardware
//
#include <hip/hip_runtime.h>
#include <hip/hip_bf16.h>

// ---------------------------------------------------------------------------
// Problem constants (from the reference)
// ---------------------------------------------------------------------------
#define BB   16
#define TT   800
#define DD   80
#define HOP  256
#define WINL 512
#define NF1  1024
#define NF2  2048
#define CCH  256
#define OUTC 222
#define GRP  8
#define ZLEN (TT * HOP)          // 204800
#define PI_F 3.14159265358979323846f

typedef __attribute__((ext_vector_type(16))) __bf16 v16bf;
typedef __attribute__((ext_vector_type(8)))  float  v8f;

// gfx1250 async global->LDS path (guarded: falls back to plain loads)
#if __has_builtin(__builtin_amdgcn_global_load_async_to_lds_b32) && \
    __has_builtin(__builtin_amdgcn_s_wait_asynccnt)
#define HAVE_ASYNC_LDS 1
typedef __attribute__((address_space(1))) int g_as_int;
typedef __attribute__((address_space(3))) int l_as_int;
#else
#define HAVE_ASYNC_LDS 0
#endif

// ---------------------------------------------------------------------------
// Prep: transpose x (B,T,D) f32 -> xT (B,D,T) bf16
// ---------------------------------------------------------------------------
__global__ void prep_xt_kernel(const float* __restrict__ x, __bf16* __restrict__ xT)
{
    int i = blockIdx.x * 256 + threadIdx.x;
    if (i >= BB * DD * TT) return;
    int b = i / (DD * TT);
    int r = i - b * DD * TT;
    int d = r / TT;
    int t = r - d * TT;
    xT[i] = (__bf16)x[(b * TT + t) * DD + d];
}

// Prep: f32 -> bf16 copy (weights)
__global__ void cvt_bf16_kernel(const float* __restrict__ src, __bf16* __restrict__ dst, int n)
{
    int i = blockIdx.x * 256 + threadIdx.x;
    if (i < n) dst[i] = (__bf16)src[i];
}

// ---------------------------------------------------------------------------
// WMMA conv1d (kernel width 3, 'same' padding).
// One wave32 computes a 16 (cout) x 64 (time) strip. The block's full input
// tile (CinG channels x 66 samples) is staged into LDS ONCE, so every global
// element is read exactly once; fragments are then assembled with ds reads.
// K loop is compile-time -> fully unrolled WMMA chain, 4 accumulators.
//   in  : (B, Cin, T) bf16 ; w : (Cout, CinG*3) bf16
//   out : bf16 (B, Cout, T)   or   f32 time-major (B, T, Cout)
// Fragment layouts per CDNA5 ISA 7.12.2 (wave32, 16-bit A, 32-bit C/D).
// ---------------------------------------------------------------------------
template <int CING, bool RELU, bool F32OUT>
__global__ void __launch_bounds__(32) conv_wmma_kernel(
    const __bf16* __restrict__ in,
    const __bf16* __restrict__ w,
    const float*  __restrict__ bias,
    __bf16* __restrict__ out_bf,
    float*  __restrict__ out_f32,
    int Cin, int Cout, int coutPerG)
{
    constexpr int Kdim   = CING * 3;
    constexpr int KSTEPS = (Kdim + 31) / 32;
    constexpr int TROW   = 68;            // 66 used, padded vs bank conflicts

    __shared__ __bf16 tile[CING * TROW];

    const int lane = threadIdx.x;          // 0..31
    const int tb   = blockIdx.x * 64;      // time strip base
    const int cb   = blockIdx.y * 16;      // out-channel tile base
    const int b    = blockIdx.z;
    const int group   = cb / coutPerG;     // tiles never cross a group
    const int cinBase = group * CING;
    const long inBase = (long)b * Cin * TT;

    const int row = lane & 15;
    const int hi  = lane >> 4;

    // prefetch the weight rows (global_prefetch_b8)
    __builtin_prefetch(&w[(cb + row) * Kdim], 0, 0);

    // ---- stage input tile: channels [cinBase, cinBase+CinG), times [tb-1, tb+65) ----
    for (int idx = lane; idx < CING * 66; idx += 32) {
        int c   = idx / 66;
        int u   = idx - c * 66;
        int tin = tb - 1 + u;
        float v = 0.f;
        if ((unsigned)tin < (unsigned)TT)
            v = (float)in[inBase + (long)(cinBase + c) * TT + tin];
        tile[c * TROW + u] = (__bf16)v;
    }
    __syncthreads();

    v8f acc[4] = {};
    #pragma unroll
    for (int s = 0; s < KSTEPS; ++s) {
        const int kb = s * 32;
        v16bf fa;
        // A fragment: weights, 16x32 bf16 (rows = out channels)
        #pragma unroll
        for (int e = 0; e < 16; ++e) {
            int kloc = (hi == 0) ? ((e < 8) ? e : e + 8)
                                 : ((e < 8) ? e + 8 : e + 16);
            int K    = kb + kloc;
            int cout = cb + row;
            float v  = 0.f;
            if (K < Kdim && cout < Cout) v = (float)w[cout * Kdim + K];
            fa[e] = (__bf16)v;
        }
        // 4 B fragments (time tiles), each 32x16 bf16, from the LDS tile
        #pragma unroll
        for (int nt = 0; nt < 4; ++nt) {
            v16bf fb;
            #pragma unroll
            for (int e = 0; e < 16; ++e) {
                int K  = kb + ((e < 8) ? (hi * 8 + e) : (hi * 8 + e + 8));
                float v = 0.f;
                if (K < Kdim) {
                    int c3 = K / 3;
                    int kk = K - c3 * 3;
                    // LDS tile index: u = (t - (tb-1)) = nt*16 + row + kk
                    v = (float)tile[c3 * TROW + nt * 16 + row + kk];
                }
                fb[e] = (__bf16)v;
            }
            acc[nt] = __builtin_amdgcn_wmma_f32_16x16x32_bf16(
                          false, fa, false, fb, (short)0, acc[nt], false, false);
        }
    }

    // C/D layout: vgpr g, lanes 0-15 -> M=g, lanes 16-31 -> M=g+8, N=lane%16
    #pragma unroll
    for (int nt = 0; nt < 4; ++nt) {
        #pragma unroll
        for (int g = 0; g < 8; ++g) {
            int cout = cb + g + 8 * hi;
            int t    = tb + nt * 16 + (lane & 15);
            if (cout < Cout && t < TT) {
                float v = acc[nt][g] + bias[cout];
                if (RELU) v = fmaxf(v, 0.f);
                if (F32OUT)
                    out_f32[((long)b * TT + t) * Cout + cout] = v;
                else
                    out_bf[((long)b * Cout + cout) * TT + t] = (__bf16)v;
            }
        }
    }
}

// ---------------------------------------------------------------------------
// In-LDS radix-2 FFT helpers (256 threads), LDS twiddle table.
//   tw[k] = exp(-i * pi * k / 1024); forward DIF uses tw, inverse DIT conjugates.
// ---------------------------------------------------------------------------
__device__ __forceinline__ void fft_dif(float* re, float* im, int n,
                                        const float* twc, const float* tws)
{
    const int tid = threadIdx.x;
    int kstep = NF2 / n;
    for (int span = n >> 1; span >= 1; span >>= 1, kstep <<= 1) {
        for (int idx = tid; idx < (n >> 1); idx += 256) {
            int j  = idx & (span - 1);
            int i0 = ((idx & ~(span - 1)) << 1) | j;
            int i1 = i0 + span;
            float ar = re[i0], ai = im[i0];
            float br = re[i1], bi = im[i1];
            re[i0] = ar + br; im[i0] = ai + bi;
            float tr = ar - br, ti = ai - bi;
            int   k  = j * kstep;
            float cs = twc[k], sn = tws[k];
            re[i1] = tr * cs - ti * sn;
            im[i1] = tr * sn + ti * cs;
        }
        __syncthreads();
    }
}

__device__ __forceinline__ void fft_dit_inv(float* re, float* im, int n,
                                            const float* twc, const float* tws)
{
    const int tid = threadIdx.x;
    int kstep = NF2 >> 1;
    for (int span = 1; span < n; span <<= 1, kstep >>= 1) {
        for (int idx = tid; idx < (n >> 1); idx += 256) {
            int j  = idx & (span - 1);
            int i0 = ((idx & ~(span - 1)) << 1) | j;
            int i1 = i0 + span;
            int   k  = j * kstep;
            float cs = twc[k], sn = -tws[k];  // conjugate for inverse
            float br = re[i1] * cs - im[i1] * sn;
            float bi = re[i1] * sn + im[i1] * cs;
            float ar = re[i0], ai = im[i0];
            re[i0] = ar + br; im[i0] = ai + bi;
            re[i1] = ar - br; im[i1] = ai - bi;
        }
        __syncthreads();
    }
    const float inv = 1.f / (float)n;
    for (int i = tid; i < n; i += 256) { re[i] *= inv; im[i] *= inv; }
    __syncthreads();
}

// ---------------------------------------------------------------------------
// Fused per-row kernel: cepstrum -> impulse -> FFT-domain correlation -> window.
// One 256-thread workgroup per (b,t) row; ~42 KB LDS (data + twiddles + frame).
// ---------------------------------------------------------------------------
__global__ void __launch_bounds__(256) fft_row_kernel(
    const float* __restrict__ ccep,   // (B, T, 222) f32
    const float* __restrict__ z,      // (B, 1, ZLEN) f32
    float* __restrict__ zw)           // (B, T, 512) f32
{
    __shared__ float sre [NF2];
    __shared__ float sim [NF2];
    __shared__ float sre2[NF2];
    __shared__ float sim2[NF2];
    __shared__ float twc [NF2 / 2];
    __shared__ float tws [NF2 / 2];
    __shared__ float sfr [WINL];      // staged noise frame

    const int bt  = blockIdx.x;
    const int b   = bt / TT;
    const int t   = bt - b * TT;
    const int tid = threadIdx.x;

    // twiddle table (built once; 4 sincos per thread)
    for (int k = tid; k < NF2 / 2; k += 256) {
        float sn, cs;
        __sincosf(-PI_F * (float)k / (float)(NF2 / 2), &sn, &cs);
        twc[k] = cs; tws[k] = sn;
    }

    // ---- stage the 512-sample noise frame (async global->LDS on gfx1250) ----
    {
        const long zoff = (long)t * HOP - (WINL / 2 - 1);   // left pad = 255
#if HAVE_ASYNC_LDS
        if (t >= 1 && t <= TT - 2) {                         // block-uniform branch
            const float* src = z + (long)b * ZLEN + zoff;
            for (int j = tid; j < WINL; j += 256) {
                __builtin_amdgcn_global_load_async_to_lds_b32(
                    (g_as_int*)(src + j),
                    (l_as_int*)&sfr[j],
                    0, 0);
            }
            __builtin_amdgcn_s_wait_asynccnt(0);
        } else
#endif
        {
            for (int j = tid; j < WINL; j += 256) {
                long zi = zoff + j;
                sfr[j] = (zi >= 0 && zi < ZLEN) ? z[(long)b * ZLEN + zi] : 0.f;
            }
        }
    }

    // ---- 1024-pt FFT of quefrency-scaled, center-padded cepstrum ----
    const int pad = (NF1 - OUTC) / 2;   // 401
    __builtin_prefetch(&ccep[((long)b * TT + t) * OUTC], 0, 0);
    for (int i = tid; i < NF1; i += 256) {
        float v = 0.f;
        int j = i - pad;
        if (j >= 0 && j < OUTC) {
            float q = (j < OUTC / 2) ? (float)(OUTC / 2 - j) : (float)(j - OUTC / 2 + 1);
            v = ccep[((long)b * TT + t) * OUTC + j] / q;
        }
        sre[i] = v; sim[i] = 0.f;
    }
    __syncthreads();
    fft_dif(sre, sim, NF1, twc, tws);

    // mag = 10^(Re/10), phase = Im   (elementwise; bit-reversed order ok)
    for (int i = tid; i < NF1; i += 256) {
        float m = __expf(sre[i] * 0.23025850929940457f);  // ln(10)/10
        float sn, cs;
        __sincosf(sim[i], &sn, &cs);
        sre[i] = m * cs; sim[i] = m * sn;
    }
    __syncthreads();
    fft_dit_inv(sre, sim, NF1, twc, tws);    // impulse response (real in sre)

    // ---- IMPF = FFT2048(imp zero-padded) ----
    for (int i = tid; i < NF2; i += 256) {
        sre2[i] = (i < NF1) ? sre[i] : 0.f;
        sim2[i] = 0.f;
    }
    __syncthreads();
    fft_dif(sre2, sim2, NF2, twc, tws);

    // ---- FRF = FFT2048(frame padded left 511 / right 512) ----
    for (int i = tid; i < NF2; i += 256) {
        int j = i - (NF1 / 2 - 1);            // 511
        sre[i] = (j >= 0 && j < WINL) ? sfr[j] : 0.f;
        sim[i] = 0.f;
    }
    __syncthreads();
    fft_dif(sre, sim, NF2, twc, tws);

    // ---- product FRF * conj(IMPF), then inverse FFT ----
    for (int i = tid; i < NF2; i += 256) {
        float ar = sre[i],  ai = sim[i];
        float br = sre2[i], bi = sim2[i];
        sre[i] = ar * br + ai * bi;
        sim[i] = ai * br - ar * bi;
    }
    __syncthreads();
    fft_dit_inv(sre, sim, NF2, twc, tws);

    // ---- Hann window, keep first 512 ----
    for (int i = tid; i < WINL; i += 256) {
        float wn = 0.5f * (1.f - __cosf(2.f * PI_F * (float)i / (float)WINL));
        zw[((long)b * TT + t) * WINL + i] = sre[i] * wn;
    }
}

// ---------------------------------------------------------------------------
// Overlap-add with roll(+1 along time) and clip
// ---------------------------------------------------------------------------
__global__ void ola_kernel(const float* __restrict__ zw, float* __restrict__ out)
{
    long i = (long)blockIdx.x * 256 + threadIdx.x;
    if (i >= (long)BB * TT * HOP) return;
    int b = (int)(i / ((long)TT * HOP));
    long r = i - (long)b * TT * HOP;
    int t = (int)(r / HOP);
    int j = (int)(r - (long)t * HOP);
    int tp = (t + TT - 1) % TT;
    float v = zw[((long)b * TT + t) * WINL + j]
            + zw[((long)b * TT + tp) * WINL + HOP + j];
    out[i] = fminf(fmaxf(v, -1.f), 1.f);
}

// ---------------------------------------------------------------------------
// Host launcher
// ---------------------------------------------------------------------------
extern "C" void kernel_launch(void* const* d_in, const int* in_sizes, int n_in,
                              void* d_out, int out_size, void* d_ws, size_t ws_size,
                              hipStream_t stream)
{
    const float* x  = (const float*)d_in[0];
    const float* z  = (const float*)d_in[1];
    const float* W1 = (const float*)d_in[2];
    const float* b1 = (const float*)d_in[3];
    const float* W2 = (const float*)d_in[4];
    const float* b2 = (const float*)d_in[5];
    const float* W3 = (const float*)d_in[6];
    const float* b3 = (const float*)d_in[7];
    const float* W4 = (const float*)d_in[8];
    const float* b4 = (const float*)d_in[9];
    float* out = (float*)d_out;

    // carve workspace
    char* p = (char*)d_ws;
    auto carve = [&](size_t bytes) {
        void* r = (void*)p;
        p += (bytes + 255) & ~(size_t)255;
        return r;
    };
    __bf16* xT   = (__bf16*)carve((size_t)BB * DD * TT * 2);
    __bf16* W1b  = (__bf16*)carve((size_t)CCH * DD * 3 * 2);
    __bf16* W2b  = (__bf16*)carve((size_t)CCH * (CCH / GRP) * 3 * 2);
    __bf16* W3b  = (__bf16*)carve((size_t)CCH * (CCH / GRP) * 3 * 2);
    __bf16* W4b  = (__bf16*)carve((size_t)OUTC * CCH * 3 * 2);
    __bf16* h1   = (__bf16*)carve((size_t)BB * CCH * TT * 2);
    __bf16* h2   = (__bf16*)carve((size_t)BB * CCH * TT * 2);
    __bf16* h3   = (__bf16*)carve((size_t)BB * CCH * TT * 2);
    float*  ccep = (float*) carve((size_t)BB * TT * OUTC * 4);
    float*  zw   = (float*) carve((size_t)BB * TT * WINL * 4);

    // --- prep ---
    {
        int n = BB * DD * TT;
        prep_xt_kernel<<<(n + 255) / 256, 256, 0, stream>>>(x, xT);
    }
    {
        int n1 = CCH * DD * 3;
        int n2 = CCH * (CCH / GRP) * 3;
        int n4 = OUTC * CCH * 3;
        cvt_bf16_kernel<<<(n1 + 255) / 256, 256, 0, stream>>>(W1, W1b, n1);
        cvt_bf16_kernel<<<(n2 + 255) / 256, 256, 0, stream>>>(W2, W2b, n2);
        cvt_bf16_kernel<<<(n2 + 255) / 256, 256, 0, stream>>>(W3, W3b, n2);
        cvt_bf16_kernel<<<(n4 + 255) / 256, 256, 0, stream>>>(W4, W4b, n4);
    }

    // --- conv stack (WMMA), 16x64 strip per wave, LDS-staged input tile ---
    dim3 blk(32, 1, 1);
    const int tStrips = (TT + 63) / 64;   // 13
    conv_wmma_kernel<DD, true, false><<<dim3(tStrips, CCH / 16, BB), blk, 0, stream>>>(
        xT, W1b, b1, h1, nullptr, DD, CCH, CCH);
    conv_wmma_kernel<CCH / GRP, true, false><<<dim3(tStrips, CCH / 16, BB), blk, 0, stream>>>(
        h1, W2b, b2, h2, nullptr, CCH, CCH, CCH / GRP);
    conv_wmma_kernel<CCH / GRP, true, false><<<dim3(tStrips, CCH / 16, BB), blk, 0, stream>>>(
        h2, W3b, b3, h3, nullptr, CCH, CCH, CCH / GRP);
    conv_wmma_kernel<CCH, false, true><<<dim3(tStrips, (OUTC + 15) / 16, BB), blk, 0, stream>>>(
        h3, W4b, b4, nullptr, ccep, CCH, OUTC, OUTC);

    // --- fused per-row spectral pipeline ---
    fft_row_kernel<<<BB * TT, 256, 0, stream>>>(ccep, z, zw);

    // --- overlap-add + clip ---
    {
        long n = (long)BB * TT * HOP;
        ola_kernel<<<(int)((n + 255) / 256), 256, 0, stream>>>(zw, out);
    }
}
